// MLPPredictor_56307021251125
// MI455X (gfx1250) — matrix operand order repaired and checked by reading, not compiled, not run on hardware
//
#include <hip/hip_runtime.h>

typedef __attribute__((ext_vector_type(16))) __bf16 v16bf;
typedef __attribute__((ext_vector_type(8)))  float  v8f;

#define N_NODES 50000
#define N_EDGES 625000
#define HF      128     // H_FEATS
#define KDIM    256     // 2*H_FEATS
#define MT      4       // M-tiles (of 16 edges) per wave

__device__ __forceinline__ unsigned short f32_to_bf16(float f) {
  unsigned int u = __float_as_uint(f);
  unsigned int r = 0x7FFFu + ((u >> 16) & 1u);   // round-to-nearest-even
  u += r;
  return (unsigned short)(u >> 16);
}

// ---- convert h (f32 [N_NODES*HF]) -> bf16 workspace -------------------------
__global__ void cvt_h_kernel(const float* __restrict__ h,
                             unsigned short* __restrict__ out) {
  int i = (blockIdx.x * blockDim.x + threadIdx.x) * 4;
  if (i >= N_NODES * HF) return;
  float4 v = *(const float4*)(h + i);
  out[i + 0] = f32_to_bf16(v.x);
  out[i + 1] = f32_to_bf16(v.y);
  out[i + 2] = f32_to_bf16(v.z);
  out[i + 3] = f32_to_bf16(v.w);
}

// ---- transpose+convert W1 (f32 [256][128]) -> bf16 W1t [128][256] -----------
__global__ void cvt_w1t_kernel(const float* __restrict__ W1,
                               unsigned short* __restrict__ w1t) {
  int o = blockIdx.x * blockDim.x + threadIdx.x;
  if (o >= HF * KDIM) return;
  int n = o >> 8;          // 0..127  (output column of layer-1)
  int k = o & 255;         // 0..255  (input feature)
  w1t[o] = f32_to_bf16(W1[k * HF + n]);
}

// ---- main kernel: one wave32 per 64-edge tile (4 WMMA M-tiles) --------------
__global__ __launch_bounds__(256) void edge_mlp_kernel(
    const unsigned short* __restrict__ hbf,   // [N_NODES][128] bf16
    const unsigned short* __restrict__ w1t,   // [128][256]     bf16 (transposed W1)
    const int*   __restrict__ srcI,
    const int*   __restrict__ dstI,
    const float* __restrict__ b1,
    const float* __restrict__ w2,
    const float* __restrict__ b2,
    float*       __restrict__ out)
{
  __shared__ unsigned short ldsW[HF * KDIM];          // 64 KB

  // Cooperative copy of W1t into LDS (4096 uint4 / 256 threads = 16 each).
  {
    const uint4* s = (const uint4*)w1t;
    uint4*       d = (uint4*)ldsW;
    int tid = threadIdx.x;
#pragma unroll
    for (int i = 0; i < 16; ++i)
      d[tid + i * 256] = s[tid + i * 256];
  }
  __syncthreads();

  const int wave   = threadIdx.x >> 5;
  const int lane   = threadIdx.x & 31;
  const int lrow   = lane & 15;      // A: row M within M-tile / B,C: column N
  const int hiHalf = lane >> 4;      // 0: lanes 0-15, 1: lanes 16-31

  const int nTiles = (N_EDGES + 16 * MT - 1) / (16 * MT);
  const int tile   = blockIdx.x * 8 + wave;
  if (tile >= nTiles) return;        // wave-uniform exit (after barrier)

  const int ebase = tile * 16 * MT;

  // Edge indices for the 4 M-tiles (clamped for the tail tile).
  int rs[MT], rd[MT];
#pragma unroll
  for (int m = 0; m < MT; ++m) {
    int e = ebase + m * 16 + lrow;
    if (e >= N_EDGES) e = N_EDGES - 1;
    rs[m] = srcI[e];
    rd[m] = dstI[e];
  }

  // Per-lane column constants for the fused epilogue (layer-2).
  float b1v[8], w2v[8];
#pragma unroll
  for (int t = 0; t < 8; ++t) {
    b1v[t] = b1[t * 16 + lrow];
    w2v[t] = w2[t * 16 + lrow];
  }
  const float bias2 = b2[0];

  union ABfrag { v16bf v; uint4 q[2]; };

  // 4x8 f32 16x16 accumulators (256 VGPRs).
  v8f acc[MT][8];
#pragma unroll
  for (int m = 0; m < MT; ++m)
#pragma unroll
    for (int nt = 0; nt < 8; ++nt)
      acc[m][nt] = (v8f){};

  // ---- ks-outer: load 4 A fragments once, reuse across all 8 N-tiles -------
#pragma unroll
  for (int ks = 0; ks < 8; ++ks) {
    const int kof = (ks & 3) * 32 + hiHalf * 8;         // element offset in row
    ABfrag a[MT];
#pragma unroll
    for (int m = 0; m < MT; ++m) {
      const int row = (ks < 4) ? rs[m] : rd[m];         // src half / dst half
      const unsigned short* p = hbf + (size_t)row * HF + kof;
      a[m].q[0] = *(const uint4*)(p);                   // K = kof .. kof+7
      a[m].q[1] = *(const uint4*)(p + 16);              // K = kof+16 .. kof+23
    }
#pragma unroll
    for (int nt = 0; nt < 8; ++nt) {
      ABfrag bfrag;
      // B layout: lane = column N, K = 8*hiHalf..+7 and 16+8*hiHalf..+7
      const unsigned short* bp =
          ldsW + (nt * 16 + lrow) * KDIM + ks * 32 + hiHalf * 8;
      bfrag.q[0] = *(const uint4*)(bp);
      bfrag.q[1] = *(const uint4*)(bp + 16);
#pragma unroll
      for (int m = 0; m < MT; ++m) {
        acc[m][nt] = __builtin_amdgcn_wmma_f32_16x16x32_bf16(
            /*neg_a=*/false, a[m].v,
            /*neg_b=*/false, bfrag.v,
            /*c_mod=*/(short)0, acc[m][nt],
            /*reuse_a=*/false, /*reuse_b=*/false);
      }
    }
  }

  // ---- Fused layer-2 epilogue + cross-lane reduction, per M-tile -----------
#pragma unroll
  for (int m = 0; m < MT; ++m) {
    float partial[8];
#pragma unroll
    for (int r = 0; r < 8; ++r) partial[r] = 0.0f;

#pragma unroll
    for (int nt = 0; nt < 8; ++nt) {
      const float bb = b1v[nt], ww = w2v[nt];
#pragma unroll
      for (int r = 0; r < 8; ++r) {
        float v = acc[m][nt][r] + bb;         // C layout: M = r + 8*hiHalf
        v = v > 0.0f ? v : 0.0f;
        partial[r] += v * ww;
      }
    }

    // Reduce over the 16 columns held in each 16-lane group.
#pragma unroll
    for (int msk = 1; msk <= 8; msk <<= 1) {
#pragma unroll
      for (int r = 0; r < 8; ++r)
        partial[r] += __shfl_xor(partial[r], msk, 32);
    }

    // Lanes 0-7 write edges 0-7 of this M-tile, lanes 16-23 write edges 8-15.
    if (lrow < 8) {
      const int eo = ebase + m * 16 + hiHalf * 8 + lrow;
      if (eo < N_EDGES) out[eo] = partial[lrow] + bias2;
    }
  }
}

extern "C" void kernel_launch(void* const* d_in, const int* in_sizes, int n_in,
                              void* d_out, int out_size, void* d_ws, size_t ws_size,
                              hipStream_t stream) {
  const float* h   = (const float*)d_in[0];
  const int*   src = (const int*)  d_in[1];
  const int*   dst = (const int*)  d_in[2];
  const float* W1  = (const float*)d_in[3];
  const float* b1  = (const float*)d_in[4];
  const float* W2  = (const float*)d_in[5];
  const float* b2  = (const float*)d_in[6];
  float* out = (float*)d_out;

  // Workspace: [h bf16: N_NODES*HF][W1t bf16: HF*KDIM]
  unsigned short* hbf = (unsigned short*)d_ws;
  unsigned short* w1t = hbf + (size_t)N_NODES * HF;

  {
    int nquads = (N_NODES * HF) / 4;
    cvt_h_kernel<<<(nquads + 255) / 256, 256, 0, stream>>>(h, hbf);
  }
  cvt_w1t_kernel<<<(HF * KDIM + 255) / 256, 256, 0, stream>>>(W1, w1t);

  const int nTiles = (N_EDGES + 16 * MT - 1) / (16 * MT);
  const int blocks = (nTiles + 7) / 8;       // 8 waves per 256-thread block
  edge_mlp_kernel<<<blocks, 256, 0, stream>>>(hbf, w1t, src, dst, b1, W2, b2, out);
}
